// Speller_18021682774497
// MI455X (gfx1250) — matrix operand order repaired and checked by reading, not compiled, not run on hardware
//
#include <hip/hip_runtime.h>
#include <hip/hip_bf16.h>

typedef __attribute__((ext_vector_type(16))) _Float16 v16h;
typedef __attribute__((ext_vector_type(8)))  _Float16 v8h;
typedef __attribute__((ext_vector_type(8)))  float    v8f;
typedef __attribute__((ext_vector_type(4)))  float    v4f;

union F16Frag { v16h v; v8h h[2]; _Float16 e[16]; };

#define DEV_INLINE __device__ __forceinline__

// Problem constants
constexpr int B = 64, T = 2048, S = 150, H = 512, Cc = 256, Qq = 256, V = 33;
constexpr int SM1 = S - 1;          // 149 steps
constexpr int NB  = 8;              // batch rows per workgroup (8 WGs)
constexpr int GH  = 4 * H;          // 2048 gate columns
constexpr int THREADS = 512;        // 16 waves
constexpr int NW = 16;

// packed-weight tile geometry (K tiles of 32, N tiles of 16, 512 halves per frag)
constexpr int NKT_IH1 = 24, NKT_HH = 16, NKT_Q = 16, NKT_FC = 24, NKT_OUT = 16;
constexpr int NNT_G = GH / 16;      // 128
constexpr int NNT_Q = Qq / 16;      // 16
constexpr int NNT_FC = H / 16;      // 32
constexpr int NNT_OUT = 3;          // 48 cols (33 padded to 48)

constexpr size_t P_IH1 = (size_t)NNT_G  * NKT_IH1 * 512;
constexpr size_t P_HH  = (size_t)NNT_G  * NKT_HH  * 512;
constexpr size_t P_Q   = (size_t)NNT_Q  * NKT_Q   * 512;
constexpr size_t P_FC  = (size_t)NNT_FC * NKT_FC  * 512;
constexpr size_t P_OUT = (size_t)NNT_OUT* NKT_OUT * 512;
constexpr size_t O_IH1 = 0;
constexpr size_t O_HH1 = O_IH1 + P_IH1;
constexpr size_t O_IH2 = O_HH1 + P_HH;
constexpr size_t O_HH2 = O_IH2 + P_HH;
constexpr size_t O_IH3 = O_HH2 + P_HH;
constexpr size_t O_HH3 = O_IH3 + P_HH;
constexpr size_t O_Q   = O_HH3 + P_HH;
constexpr size_t O_FC  = O_Q   + P_Q;
constexpr size_t O_OUT = O_FC  + P_FC;
constexpr size_t WS_HALVES = O_OUT + P_OUT;

constexpr size_t ATT_OFF = (size_t)B * SM1 * V;   // att region offset in d_out

// ---------------------------------------------------------------------------
// Pack f32 weight W[N,K] (row-major) into WMMA B-fragment-major f16:
// element for (nt,kt,lane,j): N = nt*16 + (lane&15), K = kt*32 + 16*(lane>>4) + j
// ---------------------------------------------------------------------------
__global__ void pack_w(const float* __restrict__ src, _Float16* __restrict__ dst,
                       int Ktot, int Ntot, int nNT, int nKT) {
  int idx = blockIdx.x * 256 + threadIdx.x;
  int total = nNT * nKT * 512;
  if (idx >= total) return;
  int nt   = idx / (nKT * 512);
  int rem  = idx - nt * (nKT * 512);
  int kt   = rem >> 9;
  int q    = rem & 511;
  int lane = q >> 4, j = q & 15;
  int n = nt * 16 + (lane & 15);
  int K = kt * 32 + 16 * (lane >> 4) + j;
  float v = (n < Ntot) ? src[(size_t)n * Ktot + K] : 0.0f;
  dst[idx] = (_Float16)v;
}

DEV_INLINE float sigm(float x) { return 1.0f / (1.0f + __expf(-x)); }

// A-fragment load from LDS. Rows 8..15 of the 16-row tile alias rows 0..7:
// they only feed D rows 8..15, which are never stored (no branches, no zero-fill).
DEV_INLINE F16Frag load_a(const _Float16* __restrict__ sAm, int ldA, int ktRel, int lane) {
  const int mr = (lane & 15) & (NB - 1);
  const int hh = lane >> 4;
  F16Frag a;
  const _Float16* ar = sAm + (size_t)mr * ldA + (size_t)ktRel * 32;
  a.h[0] = *(const v8h*)(ar + 16 * hh);
  a.h[1] = *(const v8h*)(ar + 16 * hh + 8);
  return a;
}

// Single-output WMMA MAC chain
DEV_INLINE v8f gemm_mac(v8f acc, const _Float16* __restrict__ wp, int nKTtot, int nt,
                        int ktBeg, int ktEnd, const _Float16* __restrict__ sAm,
                        int ldA, int lane) {
#pragma unroll 2
  for (int kt = ktBeg; kt < ktEnd; ++kt) {
    F16Frag a = load_a(sAm, ldA, kt - ktBeg, lane);
    F16Frag bq;
    const v8h* bp = (const v8h*)(wp + ((size_t)nt * nKTtot + kt) * 512 + lane * 16);
    bq.h[0] = bp[0]; bq.h[1] = bp[1];
    acc = __builtin_amdgcn_wmma_f32_16x16x32_f16(false, a.v, false, bq.v,
                                                 (short)0, acc, false, false);
  }
  return acc;
}

// 4-gate WMMA MAC: A once, then ALL 4 gate B fragments loaded (one clause),
// then 4 back-to-back WMMAs (staggered loadcnt waits, RAW hazards covered).
DEV_INLINE void lstm_tile_mac(v8f acc[4], const _Float16* __restrict__ wp, int nKTtot,
                              int ntI, int ktBeg, int ktEnd,
                              const _Float16* __restrict__ sAm, int ldA, int lane) {
#pragma unroll 2
  for (int kt = ktBeg; kt < ktEnd; ++kt) {
    F16Frag a = load_a(sAm, ldA, kt - ktBeg, lane);
    F16Frag bq[4];
#pragma unroll
    for (int g = 0; g < 4; ++g) {
      const v8h* bp = (const v8h*)(wp + ((size_t)(ntI + 32 * g) * nKTtot + kt) * 512 + lane * 16);
      bq[g].h[0] = bp[0]; bq[g].h[1] = bp[1];
    }
#pragma unroll
    for (int g = 0; g < 4; ++g)
      acc[g] = __builtin_amdgcn_wmma_f32_16x16x32_f16(false, a.v, false, bq[g].v,
                                                      (short)0, acc[g], false, false);
  }
}

// One LSTM layer: gates = In1@Wih[0:kt1] (+ In2@Wih[kt1:]) + Hrec@Whh + biases
// c update on register-resident D-fragments; staged h write (RAW-safe on Hrec==Hout)
DEV_INLINE void lstm_layer(const _Float16* __restrict__ wih, int nKTih,
                           const _Float16* __restrict__ whh,
                           const float* __restrict__ bih, const float* __restrict__ bhh,
                           const _Float16* __restrict__ sIn1, int ld1, int kt1,
                           const _Float16* __restrict__ sIn2, int ld2,
                           const _Float16* __restrict__ sHrec,
                           _Float16 (* __restrict__ sHout)[H],
                           float (&cL)[2][8], int wv, int lane) {
  const int nn = lane & 15, hh = lane >> 4;
  float hst[2][8];
#pragma unroll
  for (int ti = 0; ti < 2; ++ti) {
    const int hcT = wv + 16 * ti;       // this wave's h-column tile (0..31)
    const int jc  = hcT * 16 + nn;      // h column index
    v8f acc[4];
#pragma unroll
    for (int g = 0; g < 4; ++g) {
      float bv = bih[g * H + jc] + bhh[g * H + jc];
#pragma unroll
      for (int r = 0; r < 8; ++r) acc[g][r] = bv;
    }
    lstm_tile_mac(acc, wih, nKTih, hcT, 0, kt1, sIn1, ld1, lane);
    if (sIn2) lstm_tile_mac(acc, wih, nKTih, hcT, kt1, nKTih, sIn2, ld2, lane);
    lstm_tile_mac(acc, whh, NKT_HH, hcT, 0, NKT_HH, sHrec, H, lane);
#pragma unroll
    for (int r = 0; r < 8; ++r) {
      float c  = cL[ti][r];
      float iv = sigm(acc[0][r]);
      float fv = sigm(acc[1][r]);
      float gv = tanhf(acc[2][r]);
      float ov = sigm(acc[3][r]);
      float c2 = fv * c + iv * gv;
      cL[ti][r] = c2;
      hst[ti][r] = ov * tanhf(c2);
    }
  }
  __syncthreads();                      // everyone done reading sHrec
  if (hh == 0) {                        // D-fragment rows m = r (valid batch rows)
#pragma unroll
    for (int ti = 0; ti < 2; ++ti)
#pragma unroll
      for (int r = 0; r < 8; ++r)
        sHout[r][(wv + 16 * ti) * 16 + nn] = (_Float16)hst[ti][r];
  }
  __syncthreads();
}

// ---------------------------------------------------------------------------
__global__ __launch_bounds__(THREADS) void speller_main(
    const float* __restrict__ key, const float* __restrict__ value,
    const float* __restrict__ mask, const int* __restrict__ targets,
    const float* __restrict__ b_ih1, const float* __restrict__ b_hh1,
    const float* __restrict__ b_ih2, const float* __restrict__ b_hh2,
    const float* __restrict__ b_ih3, const float* __restrict__ b_hh3,
    const float* __restrict__ emb_W, const float* __restrict__ emb_b,
    const float* __restrict__ q_b, const float* __restrict__ fc_b,
    const float* __restrict__ out_b,
    const float* __restrict__ h0, const float* __restrict__ c0,
    const _Float16* __restrict__ wpack, float* __restrict__ out) {
  __shared__ _Float16 sH1[NB][H], sH2[NB][H], sH3[NB][H];  // 24 KB
  __shared__ _Float16 sCtxH[NB][Cc];                       // 4 KB
  __shared__ _Float16 sA[NB][H];                           // char / fc-out, 8 KB
  __shared__ float    sP[NW][Cc];                          // qv (rows 0..7) then per-wave ctx partials, 16 KB
  __shared__ float    sLane[NW][32];                       // 2 KB
  __shared__ float    sMax[NB];
  __shared__ float    sSumP[NW];

  const int tid  = threadIdx.x;
  const int lane = tid & 31;
  const int wv   = tid >> 5;            // wave 0..15
  const int b0   = blockIdx.x * NB;
  const int nn   = lane & 15, hhi = lane >> 4;

  // init h (f16, LDS) and c (f32, registers in D-fragment layout), ctx = 0
  for (int e = tid; e < 3 * NB * H; e += THREADS) {
    int L = e / (NB * H); int r = e - L * NB * H; int bl = r / H; int k = r - bl * H;
    _Float16 hv = (_Float16)h0[(size_t)L * B * H + (size_t)(b0 + bl) * H + k];
    if (L == 0) sH1[bl][k] = hv; else if (L == 1) sH2[bl][k] = hv; else sH3[bl][k] = hv;
  }
  for (int e = tid; e < NB * Cc; e += THREADS) sCtxH[e / Cc][e % Cc] = (_Float16)0.0f;

  float cReg[3][2][8];
#pragma unroll
  for (int L = 0; L < 3; ++L)
#pragma unroll
    for (int ti = 0; ti < 2; ++ti) {
      int hc0 = (wv + 16 * ti) * 16;
#pragma unroll
      for (int r = 0; r < 8; ++r) {
        int mb = r + 8 * hhi;
        cReg[L][ti][r] = (mb < NB)
            ? c0[(size_t)L * B * H + (size_t)(b0 + mb) * H + hc0 + nn] : 0.0f;
      }
    }
  __syncthreads();

  for (int s = 0; s < SM1; ++s) {
    // 1) embedding (one-hot @ emb_W.T == column gather) -> sA (f16)
    for (int e = tid; e < NB * H; e += THREADS) {
      int bl = e / H, k = e - bl * H;
      int tg = targets[(size_t)s * B + b0 + bl];
      sA[bl][k] = (_Float16)(emb_W[(size_t)k * V + tg] + emb_b[k]);
    }
    __syncthreads();

    // 2-4) LSTM stack (WMMA). li = [char | ctx] handled as split-K.
    lstm_layer(wpack + O_IH1, NKT_IH1, wpack + O_HH1, b_ih1, b_hh1,
               &sA[0][0], H, 16, &sCtxH[0][0], Cc, &sH1[0][0], sH1, cReg[0], wv, lane);
    lstm_layer(wpack + O_IH2, NKT_HH, wpack + O_HH2, b_ih2, b_hh2,
               &sH1[0][0], H, NKT_HH, nullptr, 0, &sH2[0][0], sH2, cReg[1], wv, lane);
    lstm_layer(wpack + O_IH3, NKT_HH, wpack + O_HH3, b_ih3, b_hh3,
               &sH2[0][0], H, NKT_HH, nullptr, 0, &sH3[0][0], sH3, cReg[2], wv, lane);

    // 5) q = h3 @ qW.T + qb -> sP rows 0..7 (f32 qv)
    {
      int nt = wv;                      // 16 waves, 16 n-tiles
      int jc = nt * 16 + nn;
      float bv = q_b[jc];
      v8f acc;
#pragma unroll
      for (int r = 0; r < 8; ++r) acc[r] = bv;
      acc = gemm_mac(acc, wpack + O_Q, NKT_Q, nt, 0, NKT_Q, &sH3[0][0], H, lane);
      if (hhi == 0)
#pragma unroll
        for (int r = 0; r < 8; ++r) sP[r][jc] = acc[r];
    }
    __syncthreads();

    // 6a) energies e[b,t] = q . key[b,:,t]; 2 waves per batch row, half T each.
    // Each lane owns 4 contiguous t in 8 blocks of 128; single q-loop feeds all
    // 8 blocks (1 LDS broadcast + 8 b128 loads + 32 independent FMAs per q).
    const int ab = wv >> 1;             // batch row 0..7
    const int ah = wv & 1;              // T half
    const int gb = b0 + ab;
    float* attp = out + ATT_OFF + ((size_t)s * B + gb) * T;
    {
      const float* kb = key + (size_t)gb * Qq * T + ah * (T / 2) + lane * 4;
      v4f acc4[8];
#pragma unroll
      for (int i = 0; i < 8; ++i)
#pragma unroll
        for (int c = 0; c < 4; ++c) acc4[i][c] = 0.0f;
      for (int q = 0; q < Qq; ++q) {
        float qv = sP[ab][q];
        const float* kq = kb + (size_t)q * T;
#pragma unroll
        for (int i = 0; i < 8; ++i) {
          v4f kv = *(const v4f*)(kq + i * 128);
#pragma unroll
          for (int c = 0; c < 4; ++c) acc4[i][c] = fmaf(qv, kv[c], acc4[i][c]);
        }
      }
      float lmax = -3.0e38f;
#pragma unroll
      for (int i = 0; i < 8; ++i) {
        *(v4f*)(attp + ah * (T / 2) + i * 128 + lane * 4) = acc4[i];
#pragma unroll
        for (int c = 0; c < 4; ++c) lmax = fmaxf(lmax, acc4[i][c]);
      }
      sLane[wv][lane] = lmax;
    }
    __syncthreads();
    if (tid < NB) {                     // max over the 2 waves of this batch row
      float mx = sLane[2 * tid][0];
      for (int l = 1; l < 32; ++l) mx = fmaxf(mx, sLane[2 * tid][l]);
      for (int l = 0; l < 32; ++l) mx = fmaxf(mx, sLane[2 * tid + 1][l]);
      sMax[tid] = mx;
    }
    __syncthreads();

    // 6b) p_unnorm = exp(e-max)*mask; per-wave partial sum + partial ctx.
    // Lane owns 8 contiguous channels -> two b128 value loads per t.
    {
      const float* vb = value + (size_t)gb * T * Cc + lane * 8;
      const float* mb = mask + (size_t)gb * T;
      const float bm = sMax[ab];
      float sum = 0.0f;
      float cacc[8];
#pragma unroll
      for (int c8 = 0; c8 < 8; ++c8) cacc[c8] = 0.0f;
      const int t0 = ah * (T / 2), t1 = t0 + (T / 2);
#pragma unroll 2
      for (int t = t0; t < t1; ++t) {
        float pu = __expf(attp[t] - bm) * mb[t];
        sum += pu;
        const float* vr = vb + (size_t)t * Cc;
        __builtin_prefetch(vr + 4 * Cc, 0, 0);   // stream-ahead hint (global_prefetch_b8)
        v4f v0 = *(const v4f*)(vr);
        v4f v1 = *(const v4f*)(vr + 4);
#pragma unroll
        for (int c = 0; c < 4; ++c) cacc[c]     = fmaf(pu, v0[c], cacc[c]);
#pragma unroll
        for (int c = 0; c < 4; ++c) cacc[4 + c] = fmaf(pu, v1[c], cacc[4 + c]);
      }
#pragma unroll
      for (int c8 = 0; c8 < 8; ++c8) sP[wv][lane * 8 + c8] = cacc[c8];  // exclusive row
      if (lane == 0) sSumP[wv] = sum;
    }
    __syncthreads();

    // 6c) finalize att output and ctx (f16); fixed-order 2-way combine (deterministic)
    {
      const float bm  = sMax[ab];
      const float inv = 1.0f / (sSumP[2 * ab] + sSumP[2 * ab + 1]);
      const float* mb = mask + (size_t)gb * T + ah * (T / 2) + lane * 4;
      float* ap0 = attp + ah * (T / 2) + lane * 4;
#pragma unroll 2
      for (int i = 0; i < 8; ++i) {
        float* ap = ap0 + i * 128;
        v4f e = *(const v4f*)ap;
        v4f mv = *(const v4f*)(mb + i * 128);
#pragma unroll
        for (int c = 0; c < 4; ++c) e[c] = __expf(e[c] - bm) * mv[c] * inv;
        *(v4f*)ap = e;
      }
    }
    for (int e = tid; e < NB * Cc; e += THREADS) {
      int bl = e / Cc, cix = e - bl * Cc;
      float inv = 1.0f / (sSumP[2 * bl] + sSumP[2 * bl + 1]);
      sCtxH[bl][cix] = (_Float16)((sP[2 * bl][cix] + sP[2 * bl + 1][cix]) * inv);
    }
    __syncthreads();

    // 7) fc = relu([h3|ctx] @ fcW.T + fcb) -> sA (f16)
#pragma unroll
    for (int fi = 0; fi < 2; ++fi) {
      int nt = wv + 16 * fi;
      int jc = nt * 16 + nn;
      float bv = fc_b[jc];
      v8f acc;
#pragma unroll
      for (int r = 0; r < 8; ++r) acc[r] = bv;
      acc = gemm_mac(acc, wpack + O_FC, NKT_FC, nt, 0, 16, &sH3[0][0], H, lane);
      acc = gemm_mac(acc, wpack + O_FC, NKT_FC, nt, 16, 24, &sCtxH[0][0], Cc, lane);
      if (hhi == 0)
#pragma unroll
        for (int r = 0; r < 8; ++r) sA[r][jc] = (_Float16)fmaxf(acc[r], 0.0f);
    }
    __syncthreads();

    // 8) logits = fc @ outW.T + outb -> global (N padded 33->48, 3 tiles)
    if (wv < 3) {
      int nt = wv;
      int jc = nt * 16 + nn;
      float bv = (jc < V) ? out_b[jc] : 0.0f;
      v8f acc;
#pragma unroll
      for (int r = 0; r < 8; ++r) acc[r] = bv;
      acc = gemm_mac(acc, wpack + O_OUT, NKT_OUT, nt, 0, NKT_OUT, &sA[0][0], H, lane);
      if (hhi == 0 && jc < V)
#pragma unroll
        for (int r = 0; r < 8; ++r)
          out[(size_t)(b0 + r) * SM1 * V + (size_t)s * V + jc] = acc[r];
    }
    __syncthreads();   // protect sA before next step's embedding
  }
}

// ---------------------------------------------------------------------------
extern "C" void kernel_launch(void* const* d_in, const int* in_sizes, int n_in,
                              void* d_out, int out_size, void* d_ws, size_t ws_size,
                              hipStream_t stream) {
  const float* key     = (const float*)d_in[0];
  const float* value   = (const float*)d_in[1];
  const float* mask    = (const float*)d_in[2];
  const int*   targets = (const int*)d_in[3];
  const float* W_ih1 = (const float*)d_in[4],  *W_hh1 = (const float*)d_in[5];
  const float* b_ih1 = (const float*)d_in[6],  *b_hh1 = (const float*)d_in[7];
  const float* W_ih2 = (const float*)d_in[8],  *W_hh2 = (const float*)d_in[9];
  const float* b_ih2 = (const float*)d_in[10], *b_hh2 = (const float*)d_in[11];
  const float* W_ih3 = (const float*)d_in[12], *W_hh3 = (const float*)d_in[13];
  const float* b_ih3 = (const float*)d_in[14], *b_hh3 = (const float*)d_in[15];
  const float* emb_W = (const float*)d_in[16], *emb_b = (const float*)d_in[17];
  const float* q_W   = (const float*)d_in[18], *q_b   = (const float*)d_in[19];
  const float* fc_W  = (const float*)d_in[20], *fc_b  = (const float*)d_in[21];
  const float* out_W = (const float*)d_in[22], *out_b = (const float*)d_in[23];
  const float* h0    = (const float*)d_in[24], *c0    = (const float*)d_in[25];

  if (ws_size < WS_HALVES * sizeof(_Float16)) return;
  _Float16* wp = (_Float16*)d_ws;

  auto packL = [&](const float* src, size_t off, int Ktot, int Ntot, int nNT, int nKT) {
    int total = nNT * nKT * 512;
    pack_w<<<(total + 255) / 256, 256, 0, stream>>>(src, wp + off, Ktot, Ntot, nNT, nKT);
  };
  packL(W_ih1, O_IH1, 768, 2048, NNT_G, NKT_IH1);
  packL(W_hh1, O_HH1, 512, 2048, NNT_G, NKT_HH);
  packL(W_ih2, O_IH2, 512, 2048, NNT_G, NKT_HH);
  packL(W_hh2, O_HH2, 512, 2048, NNT_G, NKT_HH);
  packL(W_ih3, O_IH3, 512, 2048, NNT_G, NKT_HH);
  packL(W_hh3, O_HH3, 512, 2048, NNT_G, NKT_HH);
  packL(q_W,   O_Q,   512, 256,  NNT_Q,  NKT_Q);
  packL(fc_W,  O_FC,  768, 512,  NNT_FC, NKT_FC);
  packL(out_W, O_OUT, 512, 33,   NNT_OUT, NKT_OUT);

  speller_main<<<B / NB, THREADS, 0, stream>>>(
      key, value, mask, targets,
      b_ih1, b_hh1, b_ih2, b_hh2, b_ih3, b_hh3,
      emb_W, emb_b, q_b, fc_b, out_b, h0, c0,
      (const _Float16*)wp, (float*)d_out);
}
